// TwikerModel_75531294868099
// MI455X (gfx1250) — matrix operand order repaired and checked by reading, not compiled, not run on hardware
//
#include <hip/hip_runtime.h>
#include <hip/hip_bf16.h>

typedef __attribute__((ext_vector_type(16))) _Float16 v16h;
typedef __attribute__((ext_vector_type(2)))  _Float16 v2h;
typedef __attribute__((ext_vector_type(8)))  float    v8f;

#define B_  2
#define H_  16
#define N_  1024
#define F_  64
#define KW  5
#define PW  2
#define G_  32
#define J_  3
#define KST 72   // K-tile LDS row stride (halves): 144B rows, 16B aligned, conflict-free
#define VST 40   // V-tile / P-buf LDS row stride (halves): 80B rows, 16B aligned

// ---------------------------------------------------------------------------
// Stage 1: token-dependent depthwise causal conv of key/value.
// conv[b,g,j,n,f] = sum_{k=0}^{KW-1-j} w[b,n,k] * kv[b,g,n+k-PW,f]
// f16 mirrors for the WMMA stage:
//   kc16[b,h,n,f]            (row-major)
//   vt16[b,h,f,n']           (transposed, n permuted within each 32-block:
//                             n' = (n & ~31) | (2*(n&15) + ((n>>4)&1)) = k')
// ---------------------------------------------------------------------------
__global__ __launch_bounds__(256) void twiker_conv(
    const int*   __restrict__ ids,    // B,N
    const float* __restrict__ key,    // B,H,N,F
    const float* __restrict__ value,  // B,H,N,F
    const float* __restrict__ emb,    // V, 2*KW
    float*       __restrict__ out,    // B,G,J,N,F
    _Float16*    __restrict__ kc16,   // B,H,N,F
    _Float16*    __restrict__ vt16)   // B,H,F,N (permuted)
{
  __shared__ int   ids_s[4];
  __shared__ float wsm[4][2 * KW];

  const int tid  = threadIdx.x;
  const int nblk = blockIdx.x % (N_ / 4);
  const int bg   = blockIdx.x / (N_ / 4);
  const int b    = bg / G_;
  const int g    = bg % G_;
  const int n0   = nblk * 4;

  if (tid < 4) ids_s[tid] = ids[b * N_ + n0 + tid];
  __syncthreads();
  if (tid < 4 * 2 * KW) {
    int nl = tid / (2 * KW), w = tid % (2 * KW);
    wsm[nl][w] = emb[(size_t)ids_s[nl] * (2 * KW) + w];
  }
  __syncthreads();

  const int nl = tid >> 6;
  const int f  = tid & 63;
  const int n  = n0 + nl;

  const float* src  = (g < H_) ? key : value;
  const float* rowp = src + (((size_t)b * H_ + (g & 15)) * N_) * F_;
  const int    base = (g < H_) ? 0 : KW;

  float s = 0.f, o0 = 0.f, o1 = 0.f, o2 = 0.f;
#pragma unroll
  for (int k = 0; k < KW; ++k) {
    int   m = n + k - PW;
    float x = (m >= 0 && m < N_) ? rowp[(size_t)m * F_ + f] : 0.f;
    s += wsm[nl][base + k] * x;
    if (k == KW - 3) o2 = s;   // j=2: last two taps dropped
    if (k == KW - 2) o1 = s;   // j=1: last tap dropped
    if (k == KW - 1) o0 = s;   // j=0: full kernel
  }
  size_t ob = ((size_t)(b * G_ + g) * J_) * (size_t)(N_ * F_) + (size_t)n * F_ + f;
  out[ob]                         = o0;
  out[ob + (size_t)(N_ * F_)]     = o1;
  out[ob + (size_t)(2 * N_ * F_)] = o2;

  const int h = g & 15;
  if (g < H_) {
    kc16[(((size_t)b * H_ + h) * N_ + n) * F_ + f] = (_Float16)o0;
  } else {
    int c     = n & 31;                                 // col within 32-block
    int nperm = (n & ~31) | (2 * (c & 15) + (c >> 4));  // k' relabel
    vt16[(((size_t)b * H_ + h) * F_ + f) * N_ + nperm] = (_Float16)o0;
  }
}

// ---------------------------------------------------------------------------
// Stage 2: streaming (flash) causal attention with twiker diagonal fixes.
// 4 waves/block; each wave owns a 16-row Q tile; 32-col K/V steps.
// Double-buffered LDS tiles with async DMA prefetch (ASYNCcnt path):
// step st+1 is in flight while the WMMAs for step st execute.
// ---------------------------------------------------------------------------
__global__ __launch_bounds__(128) void twiker_attn(
    const float*    __restrict__ q,     // B,H,N,F
    const float*    __restrict__ conv,  // B,G,J,N,F
    const _Float16* __restrict__ kc16,  // B,H,N,F
    const _Float16* __restrict__ vt16,  // B,H,F,N (permuted)
    float*          __restrict__ out)   // B,H,N,F
{
  const float scale = 0.125f;  // 1/sqrt(64)

  const int bh   = blockIdx.x;            // 0..31
  const int b    = bh >> 4, h = bh & 15;
  const int br   = blockIdx.y;            // 0..15 (row block of 64)
  const int tid  = threadIdx.x;
  const int wave = tid >> 5;
  const int lane = tid & 31;
  const int half = lane >> 4;
  const int ln   = lane & 15;
  const int r0   = br * 64 + wave * 16;

  const size_t NF = (size_t)N_ * F_;
  const float* CBK1 = conv + ((size_t)(b * G_ + h)      * J_ + 1) * NF;
  const float* CBK2 = conv + ((size_t)(b * G_ + h)      * J_ + 2) * NF;
  const float* VC   = conv + ((size_t)(b * G_ + H_ + h) * J_ + 0) * NF;
  const float* CBV1 = conv + ((size_t)(b * G_ + H_ + h) * J_ + 1) * NF;
  const float* CBV2 = conv + ((size_t)(b * G_ + H_ + h) * J_ + 2) * NF;
  const float* Q    = q    + ((size_t)b * H_ + h) * NF;
  const _Float16* KC16 = kc16 + ((size_t)b * H_ + h) * NF;
  const _Float16* VT16 = vt16 + ((size_t)b * H_ + h) * NF;

  __shared__ __align__(16) _Float16 Kt[2][32 * KST];   // [buf][col][f]
  __shared__ __align__(16) _Float16 Vt[2][F_ * VST];   // [buf][f][k']
  __shared__ __align__(16) _Float16 Pbuf[4][16 * VST]; // [M][q(k')] per wave

  const unsigned kbase = (unsigned)(size_t)(&Kt[0][0]);
  const unsigned vbase = (unsigned)(size_t)(&Vt[0][0]);

  // Issue async DMA of one 32-col step's K and V tiles into LDS buffer `bufi`.
  auto fill_tiles = [&](int step, int bufi) {
    const int cb = step * 32;
#pragma unroll
    for (int c = 0; c < 2; ++c) {
      int i = tid + 128 * c;
      {  // K tile: 32 cols x 64 f  (8 x 16B chunks per col row)
        int col = i >> 3, seg = i & 7;
        unsigned long long ga =
            (unsigned long long)(size_t)(KC16 + (size_t)(cb + col) * F_ + seg * 8);
        unsigned la = kbase + (unsigned)(bufi * (32 * KST * 2)) +
                      (unsigned)(col * (KST * 2) + seg * 16);
        asm volatile("global_load_async_to_lds_b128 %0, %1, off"
                     :: "v"(la), "v"(ga) : "memory");
      }
      {  // V tile: 64 f x 32 k'  (4 x 16B chunks per f row)
        int f = i >> 2, seg = i & 3;
        unsigned long long ga =
            (unsigned long long)(size_t)(VT16 + (size_t)f * N_ + cb + seg * 8);
        unsigned la = vbase + (unsigned)(bufi * (F_ * VST * 2)) +
                      (unsigned)(f * (VST * 2) + seg * 16);
        asm volatile("global_load_async_to_lds_b128 %0, %1, off"
                     :: "v"(la), "v"(ga) : "memory");
      }
    }
  };

  // ---- Q tile into WMMA A-layout f16 registers, scale folded in ----
  v16h qa[2];
  {
    const int row = r0 + ln;
#pragma unroll
    for (int c = 0; c < 2; ++c) {
#pragma unroll
      for (int v = 0; v < 8; ++v) {
        int kk = 16 * half + 2 * v;
        int f  = 32 * c + kk;
        qa[c][2 * v]     = (_Float16)(Q[(size_t)row * F_ + f]     * scale);
        qa[c][2 * v + 1] = (_Float16)(Q[(size_t)row * F_ + f + 1] * scale);
      }
    }
  }

  // ---- exact f32 diagonal replacement scores (one dot per lane) ----
  float dval;
  {
    const int n = r0 + ln;
    if (half == 0) {                 // d0: scores[n,n] = scale * q[n].cbk2[n]
      const float4* qq = (const float4*)(Q + (size_t)n * F_);
      const float4* kk = (const float4*)(CBK2 + (size_t)n * F_);
      float s = 0.f;
      for (int i = 0; i < F_ / 4; ++i) {
        float4 a = qq[i], c = kk[i];
        s += a.x * c.x + a.y * c.y + a.z * c.z + a.w * c.w;
      }
      dval = s * scale;
    } else {                          // d1: scores[n,n-1] = scale * q[n].cbk1[n-1]
      if (n >= 1) {
        const float4* qq = (const float4*)(Q + (size_t)n * F_);
        const float4* kk = (const float4*)(CBK1 + (size_t)(n - 1) * F_);
        float s = 0.f;
        for (int i = 0; i < F_ / 4; ++i) {
          float4 a = qq[i], c = kk[i];
          s += a.x * c.x + a.y * c.y + a.z * c.z + a.w * c.w;
        }
        dval = s * scale;
      } else dval = -1e30f;
    }
  }
  float d0v[8], d1v[8];
#pragma unroll
  for (int v = 0; v < 8; ++v) {
    int M = v + 8 * half;
    d0v[v] = __shfl(dval, M, 32);
    d1v[v] = __shfl(dval, 16 + M, 32);
  }

  // ---- online-softmax state ----
  float mrow[8], lrow[8];
  v8f   oc[4];
#pragma unroll
  for (int v = 0; v < 8; ++v) { mrow[v] = -1e30f; lrow[v] = 0.f; }
  {
    v8f z = {};
#pragma unroll
    for (int cc = 0; cc < 4; ++cc) oc[cc] = z;
  }

  // P-store column position: k' itself so A-fragments are contiguous
  const int kp2  = 2 * ln;
  const int qpos = kp2;

  const int nsteps = 2 * br + 2;  // causal: cols [0, 64*br+64) in 32-col steps

  // ---- pipeline prologue: fill buffer 0 with step 0 ----
  fill_tiles(0, 0);
  asm volatile("s_wait_asynccnt 0" ::: "memory");
  __syncthreads();

  for (int st = 0; st < nsteps; ++st) {
    const int cbase = st * 32;
    const int buf   = st & 1;
    const _Float16* KtB = &Kt[buf][0];
    const _Float16* VtB = &Vt[buf][0];

    // ---- prefetch next step into the other buffer (uniform branch) ----
    if (st + 1 < nsteps) fill_tiles(st + 1, 1 - buf);

    // ---- S = (Q*scale) . Kc^T, two 16-col subtiles ----
    float p[2][8];
#pragma unroll
    for (int s2 = 0; s2 < 2; ++s2) {
      v8f c = {};
#pragma unroll
      for (int c2 = 0; c2 < 2; ++c2) {
        v16h kb;
        __builtin_memcpy(&kb, &KtB[(16 * s2 + ln) * KST + 32 * c2 + 16 * half],
                         sizeof(kb));
        c = __builtin_amdgcn_wmma_f32_16x16x32_f16(false, qa[c2], false, kb,
                                                   (short)0, c, false, false);
      }
      // causal mask + diagonal score replacement, in C-layout
#pragma unroll
      for (int v = 0; v < 8; ++v) {
        int   row_g = r0 + v + 8 * half;
        int   col_g = cbase + 16 * s2 + ln;
        float sv    = c[v];
        if      (col_g >  row_g)     sv = -1e30f;
        else if (col_g == row_g)     sv = d0v[v];
        else if (col_g == row_g - 1) sv = d1v[v];
        p[s2][v] = sv;
      }
    }

    // ---- softmax state update over the 32 new columns ----
    float alpha[8];
#pragma unroll
    for (int v = 0; v < 8; ++v) {
      float x = fmaxf(p[0][v], p[1][v]);
      for (int o = 8; o; o >>= 1) x = fmaxf(x, __shfl_xor(x, o, 16));
      float mn = fmaxf(mrow[v], x);
      alpha[v] = __expf(mrow[v] - mn);
      mrow[v]  = mn;
    }
    float rs[8];
#pragma unroll
    for (int v = 0; v < 8; ++v) rs[v] = 0.f;
#pragma unroll
    for (int s2 = 0; s2 < 2; ++s2)
#pragma unroll
      for (int v = 0; v < 8; ++v) {
        float e = __expf(p[s2][v] - mrow[v]);
        p[s2][v] = e;
        rs[v] += e;
      }
#pragma unroll
    for (int v = 0; v < 8; ++v) {
      float x = rs[v];
      for (int o = 8; o; o >>= 1) x += __shfl_xor(x, o, 16);
      lrow[v] = lrow[v] * alpha[v] + x;
    }
#pragma unroll
    for (int cc = 0; cc < 4; ++cc)
#pragma unroll
      for (int v = 0; v < 8; ++v) oc[cc][v] *= alpha[v];

    // ---- P: C-layout -> A-layout via per-wave LDS staging ----
#pragma unroll
    for (int v = 0; v < 8; ++v) {
      v2h pr; pr[0] = (_Float16)p[0][v]; pr[1] = (_Float16)p[1][v];
      *(v2h*)(&Pbuf[wave][(v + 8 * half) * VST + qpos]) = pr;
    }
    asm volatile("s_wait_dscnt 0" ::: "memory");  // same-wave LDS write->read
    v16h pa;
    __builtin_memcpy(&pa, &Pbuf[wave][ln * VST + 16 * half], sizeof(pa));

    // ---- O += P . Vc  (K=32 in k' ordering, matching Vt) ----
#pragma unroll
    for (int cc = 0; cc < 4; ++cc) {
      v16h vb;
      __builtin_memcpy(&vb, &VtB[(16 * cc + ln) * VST + 16 * half], sizeof(vb));
      oc[cc] = __builtin_amdgcn_wmma_f32_16x16x32_f16(false, pa, false, vb,
                                                      (short)0, oc[cc], false, false);
    }

    // ---- pipeline epilogue for this step: prefetch done + tiles consumed ----
    asm volatile("s_wait_asynccnt 0" ::: "memory");
    __syncthreads();
  }

  // ---- epilogue: normalize + diagonal value corrections ----
  float p0r[8], p1r[8], linv[8];
#pragma unroll
  for (int v = 0; v < 8; ++v) {
    float li = 1.0f / lrow[v];
    linv[v] = li;
    p0r[v]  = __expf(d0v[v] - mrow[v]) * li;  // probs[n,n]
    p1r[v]  = __expf(d1v[v] - mrow[v]) * li;  // probs[n,n-1] (0 for n==0)
  }
#pragma unroll
  for (int cc = 0; cc < 4; ++cc)
#pragma unroll
    for (int v = 0; v < 8; ++v) {
      int   row_g = r0 + v + 8 * half;
      int   f     = 16 * cc + ln;
      float res   = oc[cc][v] * linv[v];
      float vc    = VC[(size_t)row_g * F_ + f];
      res += p0r[v] * (CBV2[(size_t)row_g * F_ + f] - vc);
      if (row_g >= 1)
        res += p1r[v] * (CBV1[(size_t)(row_g - 1) * F_ + f] -
                         VC[(size_t)(row_g - 1) * F_ + f]);
      out[((size_t)b * H_ + h) * NF + (size_t)row_g * F_ + f] = res;
    }
}

// ---------------------------------------------------------------------------
extern "C" void kernel_launch(void* const* d_in, const int* in_sizes, int n_in,
                              void* d_out, int out_size, void* d_ws, size_t ws_size,
                              hipStream_t stream) {
  (void)in_sizes; (void)n_in; (void)out_size; (void)ws_size;
  const int*   ids = (const int*)d_in[0];
  const float* qry = (const float*)d_in[1];
  const float* key = (const float*)d_in[2];
  const float* val = (const float*)d_in[3];
  const float* emb = (const float*)d_in[4];

  float*    conv = (float*)d_ws;                                        // 50.3 MB
  _Float16* kc16 = (_Float16*)(conv + (size_t)B_ * G_ * J_ * N_ * F_);  // 4 MB
  _Float16* vt16 = kc16 + (size_t)B_ * H_ * N_ * F_;                    // 4 MB

  dim3 g1(B_ * G_ * (N_ / 4));
  twiker_conv<<<g1, 256, 0, stream>>>(ids, key, val, emb, conv, kc16, vt16);

  dim3 g2(B_ * H_, N_ / 64);
  twiker_attn<<<g2, 128, 0, stream>>>(qry, conv, kc16, vt16, (float*)d_out);
}
